// VisionTransformer_82806969467027
// MI455X (gfx1250) — matrix operand order repaired and checked by reading, not compiled, not run on hardware
//
#include <hip/hip_runtime.h>
#include <hip/hip_bf16.h>

typedef __attribute__((ext_vector_type(16))) __bf16 v16bf;
typedef __attribute__((ext_vector_type(4)))  __bf16 v4bf;
typedef __attribute__((ext_vector_type(8)))  float  v8f;
typedef __attribute__((ext_vector_type(4)))  unsigned v4u;
typedef __attribute__((ext_vector_type(8)))  int    v8i;
typedef __attribute__((ext_vector_type(4)))  int    v4i;

#define B_c      2
#define E_c      768
#define NH_c     12
#define DH_c     64
#define L_c      12
#define F_c      3072
#define S_c      2049
#define NPATCH_c 2048
#define KPATCH_c 4096
#define MROWS_c  (B_c * S_c)   // 4098

__device__ __forceinline__ v8f vzero8() {
  v8f z;
  for (int i = 0; i < 8; ++i) z[i] = 0.0f;
  return z;
}
__device__ __forceinline__ v4bf pack4(float4 f) {
  v4bf p;
  p[0] = (__bf16)f.x; p[1] = (__bf16)f.y; p[2] = (__bf16)f.z; p[3] = (__bf16)f.w;
  return p;
}

// ---------------------------------------------------------------------------
// Weight pre-conversion (once per use, amortized over all M-tiles):
//  wconv_t: fp32 [K,N] row-major -> bf16 [N,K] row-major (transpose+cast)
//  wconv  : fp32 [N,K] row-major -> bf16 [N,K]           (cast only; conv_w)
// ---------------------------------------------------------------------------
__global__ void wconv_t_kernel(const float* __restrict__ in, __bf16* __restrict__ outT,
                               int K, int N) {
  size_t idx = (size_t)blockIdx.x * 256 + threadIdx.x;
  if (idx >= (size_t)K * N) return;
  int n = (int)(idx % N);
  int kk = (int)(idx / N);           // coalesced read along N
  outT[(size_t)n * K + kk] = (__bf16)in[idx];
}
__global__ void wconv_kernel(const float* __restrict__ in, __bf16* __restrict__ outO,
                             size_t total) {
  size_t idx = (size_t)blockIdx.x * 256 + threadIdx.x;
  if (idx < total) outO[idx] = (__bf16)in[idx];
}

// ---------------------------------------------------------------------------
// im2col for 16^3 patches of the (128,256,256) volume, one batch at a time.
// ---------------------------------------------------------------------------
__global__ void im2col_kernel(const float* __restrict__ x, float* __restrict__ col, int b) {
  int i4 = blockIdx.x * 256 + threadIdx.x;
  if (i4 >= NPATCH_c * KPATCH_c / 4) return;
  int idx = i4 * 4;
  int e = idx & (KPATCH_c - 1);
  int p = idx >> 12;
  int dx = e & 15, dy = (e >> 4) & 15, dz = e >> 8;
  int pw = p & 15, ph = (p >> 4) & 15, pd = p >> 8;
  size_t zi = (size_t)pd * 16 + dz;
  size_t yi = (size_t)ph * 16 + dy;
  size_t xi = (size_t)pw * 16 + dx;
  float4 f = *(const float4*)(x + (((size_t)b * 128 + zi) * 256 + yi) * 256 + xi);
  *(float4*)(col + idx) = f;
}

// h[b,s,:] = (s==0 ? cls : xp[b,s-1,:]) + pos[s,:]
__global__ void build_h_kernel(const float* __restrict__ xp, const float* __restrict__ cls,
                               const float* __restrict__ pos, float* __restrict__ h) {
  size_t idx = (size_t)blockIdx.x * 256 + threadIdx.x;
  if (idx >= (size_t)MROWS_c * E_c) return;
  int e = (int)(idx % E_c);
  size_t rs = idx / E_c;
  int s = (int)(rs % S_c);
  int b = (int)(rs / S_c);
  float v = (s == 0) ? cls[e] : xp[((size_t)b * NPATCH_c + (s - 1)) * E_c + e];
  h[idx] = v + pos[(size_t)s * E_c + e];
}

// ---------------------------------------------------------------------------
// LayerNorm over E=768, one row per block.
// ---------------------------------------------------------------------------
__global__ __launch_bounds__(256) void layernorm_kernel(const float* __restrict__ x,
                                                        const float* __restrict__ g,
                                                        const float* __restrict__ bta,
                                                        float* __restrict__ y) {
  int row = blockIdx.x;
  const float* xr = x + (size_t)row * E_c;
  __shared__ float red[256];
  int t = threadIdx.x;
  float s = 0.f;
  for (int i = t; i < E_c; i += 256) s += xr[i];
  red[t] = s; __syncthreads();
  for (int o = 128; o > 0; o >>= 1) { if (t < o) red[t] += red[t + o]; __syncthreads(); }
  float mean = red[0] * (1.0f / E_c);
  __syncthreads();
  float v = 0.f;
  for (int i = t; i < E_c; i += 256) { float d = xr[i] - mean; v += d * d; }
  red[t] = v; __syncthreads();
  for (int o = 128; o > 0; o >>= 1) { if (t < o) red[t] += red[t + o]; __syncthreads(); }
  float rstd = rsqrtf(red[0] * (1.0f / E_c) + 1e-5f);
  float* yr = y + (size_t)row * E_c;
  for (int i = t; i < E_c; i += 256) yr[i] = (xr[i] - mean) * rstd * g[i] + bta[i];
}

// ---------------------------------------------------------------------------
// WMMA bf16 GEMM: C = A[M,K](fp32) * WtT[N,K](bf16) (+bias)(+GELU)(+resid)
// FUSE bit0 = bias, bit1 = exact GELU, bit2 = residual add.
// Tile 128x128, BK=32; 256 threads = 8 waves (4x2); wave owns 32x64 = 2x4 accs.
// B tile (128 rows x 32 k = 8KB of bf16) is fetched by the TENSOR DATA MOVER:
// a 2D D# descriptor is built per K-step and issued by one wave with
// tensor_load_to_lds; completion via s_wait_tensorcnt. The tile lands
// row-major in LDS, exactly the BsT[n][k] layout the fragments read
// (contiguous 32B runs -> ds_load_b128 pairs).
// ---------------------------------------------------------------------------
#define TM_g 128
#define TN_g 128
#define TK_g 32
#define LDSOFF_BST 8192   // BsT follows As[128][32] (8KB) in LDS

template <int FUSE>
__global__ __launch_bounds__(256) void gemm_kernel(
    const float* __restrict__ A, const __bf16* __restrict__ WtT,
    const float* __restrict__ bias, const float* __restrict__ resid,
    float* __restrict__ C, int M, int N, int K)
{
  __shared__ __bf16 As [TM_g][TK_g];
  __shared__ __bf16 BsT[TN_g][TK_g];
  const int tid  = threadIdx.x;
  const int lane = tid & 31;
  const int wave = tid >> 5;
  const int wm   = wave & 3;   // 4 row groups of 32
  const int wn   = wave >> 2;  // 2 col groups of 64
  const int bm   = blockIdx.x * TM_g;
  const int bn   = blockIdx.y * TN_g;

  v8f acc[2][4];
  for (int i = 0; i < 2; ++i)
    for (int j = 0; j < 4; ++j) acc[i][j] = vzero8();

  for (int k0 = 0; k0 < K; k0 += TK_g) {
    // --- issue TDM load of the B tile (one wave; overlaps A staging) ------
    if (tid == 0) {
      unsigned long long ga =
          (unsigned long long)(const void*)(WtT + (size_t)bn * K + k0);
      v4u g0;
      g0[0] = 1u;                                   // count=1 valid descriptor
      g0[1] = LDSOFF_BST;                           // lds_addr
      g0[2] = (unsigned)ga;                         // global_addr[31:0]
      g0[3] = (unsigned)(ga >> 32) | (2u << 30);    // global_addr[56:32] | type=2
      v8i g1;
      g1[0] = (int)(1u << 16);                      // data_size=1 (2 bytes)
      g1[1] = (int)(((unsigned)K & 0xFFFFu) << 16); // tensor_dim0[15:0]
      g1[2] = (int)((((unsigned)K >> 16) & 0xFFFFu) |
                    (((unsigned)N & 0xFFFFu) << 16));      // dim0 hi | dim1 lo
      g1[3] = (int)((((unsigned)N >> 16) & 0xFFFFu) |
                    ((unsigned)TK_g << 16));               // dim1 hi | tile_dim0
      g1[4] = (int)TN_g;                            // tile_dim1 (tile_dim2=0)
      g1[5] = (int)K;                               // tensor_dim0_stride lo
      g1[6] = 0;                                    // stride hi | dim1_stride lo
      g1[7] = 0;
      v4i gz4;
      gz4[0] = 0; gz4[1] = 0; gz4[2] = 0; gz4[3] = 0;
      v8i gz8;
      for (int i = 0; i < 8; ++i) gz8[i] = 0;
      // 6-arg toolchain form: (g0, g1, g2, g3, g4, cpol)
      __builtin_amdgcn_tensor_load_to_lds(g0, g1, gz4, gz4, gz8, 0);
    }
    // --- stage A: float4 global loads, packed v4bf LDS stores -------------
    for (int i = tid; i < TM_g * TK_g / 4; i += 256) {
      int r  = i >> 3;
      int c4 = (i & 7) * 4;
      int gr = bm + r;
      float4 f = make_float4(0.f, 0.f, 0.f, 0.f);
      if (gr < M) f = *(const float4*)(A + (size_t)gr * K + (k0 + c4));
      *(v4bf*)&As[r][c4] = pack4(f);
    }
    if (k0 + TK_g < K) {  // gfx1250 global_prefetch_b8 of next A tile
      int pr = bm + (tid >> 1);
      if (pr >= M) pr = M - 1;
      __builtin_prefetch(A + (size_t)pr * K + (k0 + TK_g), 0, 1);
    }
    if (tid == 0) __builtin_amdgcn_s_wait_tensorcnt(0);
    __syncthreads();

    // fragments: A and B lane-half both select K 16-run.
    v16bf afrag[2], bfrag[4];
    const int akb = (lane >> 4) * 16;
    const int bkh = (lane >> 4) * 16;
    for (int t = 0; t < 2; ++t) {
      int arow = wm * 32 + t * 16 + (lane & 15);
      for (int i = 0; i < 16; ++i)
        afrag[t][i] = As[arow][akb + i];
    }
    for (int t = 0; t < 4; ++t) {
      int bcol = wn * 64 + t * 16 + (lane & 15);
      for (int i = 0; i < 16; ++i)
        bfrag[t][i] = BsT[bcol][bkh + i];        // contiguous -> 2x ds_load_b128
    }
    for (int i = 0; i < 2; ++i)
      for (int j = 0; j < 4; ++j)
        acc[i][j] = __builtin_amdgcn_wmma_f32_16x16x32_bf16(
            false, afrag[i], false, bfrag[j], (short)0, acc[i][j], false, false);
    __syncthreads();
  }

  // C layout: lane holds col = lane&15; row = (lane>>4)*8 + v.
  for (int i = 0; i < 2; ++i) {
    for (int j = 0; j < 4; ++j) {
      int col = bn + wn * 64 + j * 16 + (lane & 15);
      for (int v = 0; v < 8; ++v) {
        int row = bm + wm * 32 + i * 16 + ((lane >> 4) * 8) + v;
        if (row < M && col < N) {
          float val = acc[i][j][v];
          if (FUSE & 1) val += bias[col];
          if (FUSE & 2) val = 0.5f * val * (1.0f + erff(val * 0.70710678118f));
          if (FUSE & 4) val += resid[(size_t)row * N + col];
          C[(size_t)row * N + col] = val;
        }
      }
    }
  }
}

// ---------------------------------------------------------------------------
// One dilated-attention branch (m = w/r = 512 gathered positions per
// (batch, segment, head): pos = n*w + (h%r) + j*r). Flash-style: 64-query
// tile per block, 8 key tiles of 64, online softmax. 128 threads = 4 waves.
// ---------------------------------------------------------------------------
__global__ __launch_bounds__(128) void attn_branch_kernel(
    const float* __restrict__ q, const float* __restrict__ k,
    const float* __restrict__ v, float* __restrict__ obr,
    float* __restrict__ lbr, int w, int r, int nseg)
{
  __shared__ __bf16 Qs [64][64];
  __shared__ __bf16 Ks [64][64];
  __shared__ __bf16 VsT[64][64];   // [d][key]
  __shared__ float  Sc [64][64];
  __shared__ __bf16 Pb [64][64];
  __shared__ float  rowm[64], rowl[64], corr[64];

  const int tid  = threadIdx.x;
  const int lane = tid & 31;
  const int wave = tid >> 5;   // 0..3
  int bid = blockIdx.x;
  const int qt = bid & 7; bid >>= 3;
  const int hh = bid % NH_c;  bid /= NH_c;
  const int n  = bid % nseg;
  const int b  = bid / nseg;
  const int base = n * w + (hh % r);

  for (int i = tid; i < 64 * 16; i += 128) {
    int row = i >> 4;
    int d4  = (i & 15) * 4;
    int pos = base + (qt * 64 + row) * r;
    float4 f = make_float4(0.f, 0.f, 0.f, 0.f);
    if (pos < S_c) f = *(const float4*)(q + (((size_t)b * S_c + pos) * NH_c + hh) * DH_c + d4);
    *(v4bf*)&Qs[row][d4] = pack4(f);
  }
  if (tid < 64) { rowm[tid] = -1e30f; rowl[tid] = 0.0f; }
  v8f oacc[4];
  for (int i = 0; i < 4; ++i) oacc[i] = vzero8();
  __syncthreads();

  const int akb = (lane >> 4) * 16;
  const int bkh = (lane >> 4) * 16;

  for (int kt = 0; kt < 8; ++kt) {
    for (int i = tid; i < 64 * 16; i += 128) {
      int row = i >> 4;
      int d4  = (i & 15) * 4;
      int pos = base + (kt * 64 + row) * r;
      float4 fk = make_float4(0.f, 0.f, 0.f, 0.f);
      float4 fv = fk;
      if (pos < S_c) {
        size_t gi = (((size_t)b * S_c + pos) * NH_c + hh) * DH_c + d4;
        fk = *(const float4*)(k + gi);
        fv = *(const float4*)(v + gi);
      }
      *(v4bf*)&Ks[row][d4] = pack4(fk);
      VsT[d4 + 0][row] = (__bf16)fv.x;
      VsT[d4 + 1][row] = (__bf16)fv.y;
      VsT[d4 + 2][row] = (__bf16)fv.z;
      VsT[d4 + 3][row] = (__bf16)fv.w;
    }
    __syncthreads();

    // S = Q * K^T : d is the WMMA K-dim (2 steps of 32), 4 key-column tiles.
    v8f sacc[4];
    for (int i = 0; i < 4; ++i) sacc[i] = vzero8();
    for (int ks = 0; ks < 2; ++ks) {
      v16bf aq;
      int arow = wave * 16 + (lane & 15);
      for (int i = 0; i < 16; ++i)
        aq[i] = Qs[arow][ks * 32 + akb + i];
      for (int ct = 0; ct < 4; ++ct) {
        v16bf bkf;
        int bcol = ct * 16 + (lane & 15);
        for (int i = 0; i < 16; ++i)
          bkf[i] = Ks[bcol][ks * 32 + bkh + i];   // contiguous
        sacc[ct] = __builtin_amdgcn_wmma_f32_16x16x32_bf16(
            false, aq, false, bkf, (short)0, sacc[ct], false, false);
      }
    }
    for (int ct = 0; ct < 4; ++ct) {
      int col = ct * 16 + (lane & 15);
      int pos = base + (kt * 64 + col) * r;
      float msk = (pos < S_c) ? 0.0f : -1e9f;
      for (int vv = 0; vv < 8; ++vv) {
        int row = wave * 16 + ((lane >> 4) * 8) + vv;
        Sc[row][col] = sacc[ct][vv] * 0.125f + msk;  // d^-0.5 = 1/8
      }
    }
    __syncthreads();

    if (tid < 64) {
      float mold = rowm[tid];
      float mnew = mold;
      for (int c = 0; c < 64; ++c) mnew = fmaxf(mnew, Sc[tid][c]);
      float cf = __expf(mold - mnew);
      float lsum = 0.0f;
      for (int c = 0; c < 64; c += 4) {
        float p0 = __expf(Sc[tid][c + 0] - mnew);
        float p1 = __expf(Sc[tid][c + 1] - mnew);
        float p2 = __expf(Sc[tid][c + 2] - mnew);
        float p3 = __expf(Sc[tid][c + 3] - mnew);
        v4bf pp; pp[0] = (__bf16)p0; pp[1] = (__bf16)p1; pp[2] = (__bf16)p2; pp[3] = (__bf16)p3;
        *(v4bf*)&Pb[tid][c] = pp;
        lsum += p0 + p1 + p2 + p3;
      }
      rowm[tid] = mnew;
      rowl[tid] = rowl[tid] * cf + lsum;
      corr[tid] = cf;
    }
    __syncthreads();

    for (int dt = 0; dt < 4; ++dt)
      for (int vv = 0; vv < 8; ++vv)
        oacc[dt][vv] *= corr[wave * 16 + ((lane >> 4) * 8) + vv];
    for (int ks = 0; ks < 2; ++ks) {
      v16bf ap;
      int arow = wave * 16 + (lane & 15);
      for (int i = 0; i < 16; ++i)
        ap[i] = Pb[arow][ks * 32 + akb + i];
      for (int dt = 0; dt < 4; ++dt) {
        v16bf bvf;
        int bcol = dt * 16 + (lane & 15);
        for (int i = 0; i < 16; ++i)
          bvf[i] = VsT[bcol][ks * 32 + bkh + i];  // contiguous
        oacc[dt] = __builtin_amdgcn_wmma_f32_16x16x32_bf16(
            false, ap, false, bvf, (short)0, oacc[dt], false, false);
      }
    }
    __syncthreads();
  }

  for (int dt = 0; dt < 4; ++dt) {
    int d = dt * 16 + (lane & 15);
    for (int vv = 0; vv < 8; ++vv) {
      int row = wave * 16 + ((lane >> 4) * 8) + vv;
      int pos = base + (qt * 64 + row) * r;
      if (pos < S_c)
        obr[(((size_t)b * S_c + pos) * NH_c + hh) * DH_c + d] = oacc[dt][vv] / rowl[row];
    }
  }
  if (tid < 64) {
    int pos = base + (qt * 64 + tid) * r;
    if (pos < S_c)
      lbr[((size_t)b * S_c + pos) * NH_c + hh] = rowm[tid] + __logf(rowl[tid]);
  }
}

// ---------------------------------------------------------------------------
// Combine 3 branches with softmax over per-(b,s,h) LSEs; branch validity is
// (s % w) % r == h % r (branch 0, r=1, is always valid).
// ---------------------------------------------------------------------------
__global__ void combine_kernel(const float* __restrict__ o0, const float* __restrict__ o1,
                               const float* __restrict__ o2, const float* __restrict__ l0,
                               const float* __restrict__ l1, const float* __restrict__ l2,
                               float* __restrict__ out)
{
  size_t idx = (size_t)blockIdx.x * 256 + threadIdx.x;
  if (idx >= (size_t)MROWS_c * E_c) return;
  int hh = (int)((idx >> 6) % NH_c);
  size_t sflat = idx / (64 * NH_c);  // b*S + s
  int s = (int)(sflat % S_c);
  const int wsz[3] = {512, 1024, 2048};
  const int rsz[3] = {1, 2, 4};
  const float* lp[3] = {l0, l1, l2};
  const float* op[3] = {o0, o1, o2};
  float lse[3]; bool val[3];
  float mx = -1e30f;
  for (int br = 0; br < 3; ++br) {
    val[br] = ((s % wsz[br]) % rsz[br]) == (hh % rsz[br]);
    if (val[br]) { lse[br] = lp[br][sflat * NH_c + hh]; mx = fmaxf(mx, lse[br]); }
  }
  float wsum = 0.f, acc = 0.f;
  for (int br = 0; br < 3; ++br) {
    if (val[br]) {
      float e = __expf(lse[br] - mx);
      wsum += e;
      acc += e * op[br][idx];
    }
  }
  out[idx] = acc / wsum;
}

__global__ void extract_kernel(const float* __restrict__ y, float* __restrict__ out) {
  int i = blockIdx.x * 256 + threadIdx.x;
  if (i < B_c * E_c) {
    int b = i / E_c, e = i % E_c;
    out[i] = y[(size_t)b * S_c * E_c + e];
  }
}

// ---------------------------------------------------------------------------
extern "C" void kernel_launch(void* const* d_in, const int* in_sizes, int n_in,
                              void* d_out, int out_size, void* d_ws, size_t ws_size,
                              hipStream_t stream)
{
  const float* x       = (const float*)d_in[0];
  const float* conv_w  = (const float*)d_in[1];
  const float* conv_b  = (const float*)d_in[2];
  const float* cls_tok = (const float*)d_in[3];
  const float* pos_emb = (const float*)d_in[4];
  const float* ln1_g   = (const float*)d_in[5];
  const float* ln1_b   = (const float*)d_in[6];
  const float* wq      = (const float*)d_in[7];
  const float* bq      = (const float*)d_in[8];
  const float* wk      = (const float*)d_in[9];
  const float* bk      = (const float*)d_in[10];
  const float* wv      = (const float*)d_in[11];
  const float* bv      = (const float*)d_in[12];
  const float* wo      = (const float*)d_in[13];
  const float* bo      = (const float*)d_in[14];
  const float* ln2_g   = (const float*)d_in[15];
  const float* ln2_b   = (const float*)d_in[16];
  const float* w1      = (const float*)d_in[17];
  const float* b1      = (const float*)d_in[18];
  const float* w2      = (const float*)d_in[19];
  const float* b2      = (const float*)d_in[20];
  const float* nf_g    = (const float*)d_in[21];
  const float* nf_b    = (const float*)d_in[22];
  float* out = (float*)d_out;
  (void)in_sizes; (void)n_in; (void)out_size; (void)ws_size;

  size_t off = 0;
  float* wsf = (float*)d_ws;
  auto take = [&](size_t nelts) { float* p = wsf + off; off += nelts; return p; };
  const size_t HE = (size_t)MROWS_c * E_c;
  float* h   = take(HE);
  float* y   = take(HE);
  float* qb  = take(HE);
  float* kb  = take(HE);
  float* vb  = take(HE);
  float* ao  = take(HE);
  float* o0  = take(HE);
  float* o1  = take(HE);
  float* o2  = take(HE);
  float* l0  = take((size_t)MROWS_c * NH_c);
  float* l1  = take((size_t)MROWS_c * NH_c);
  float* l2  = take((size_t)MROWS_c * NH_c);
  float* mid = take((size_t)MROWS_c * F_c);

  // bf16 weight region (reused every layer; conv slot used once up front)
  __bf16* bwbase = (__bf16*)(wsf + off);
  size_t boff = 0;
  auto takeb = [&](size_t nelts) { __bf16* p = bwbase + boff; boff += nelts; return p; };
  __bf16* wtq = takeb((size_t)E_c * E_c);
  __bf16* wtk = takeb((size_t)E_c * E_c);
  __bf16* wtv = takeb((size_t)E_c * E_c);
  __bf16* wto = takeb((size_t)E_c * E_c);
  __bf16* wt1 = takeb((size_t)E_c * F_c);   // [F][E]
  __bf16* wt2 = takeb((size_t)E_c * F_c);   // [E][F]
  __bf16* wtc = takeb((size_t)E_c * KPATCH_c);

  // phase-A overlays (only live before the layer loop)
  float* col = mid;  // 2048*4096 floats  (< MROWS*F)
  float* xp  = qb;   // B*2048*768 floats (fits qb)

  // ---- patch embedding: conv weight cast, per-batch im2col + WMMA GEMM
  {
    size_t tot = (size_t)E_c * KPATCH_c;
    wconv_kernel<<<(unsigned)((tot + 255) / 256), 256, 0, stream>>>(conv_w, wtc, tot);
  }
  for (int b = 0; b < B_c; ++b) {
    int total4 = NPATCH_c * KPATCH_c / 4;
    im2col_kernel<<<(total4 + 255) / 256, 256, 0, stream>>>(x, col, b);
    dim3 g((NPATCH_c + 127) / 128, (E_c + 127) / 128);
    gemm_kernel<1><<<g, 256, 0, stream>>>(
        col, wtc, conv_b, nullptr,
        xp + (size_t)b * NPATCH_c * E_c, NPATCH_c, E_c, KPATCH_c);
  }
  {
    size_t total = (size_t)MROWS_c * E_c;
    build_h_kernel<<<(unsigned)((total + 255) / 256), 256, 0, stream>>>(xp, cls_tok, pos_emb, h);
  }

  const int wsz[3]  = {512, 1024, 2048};
  const int rsz[3]  = {1, 2, 4};
  const int nsg[3]  = {5, 3, 2};   // ceil(2049 / w)
  float* obrs[3] = {o0, o1, o2};
  float* lbrs[3] = {l0, l1, l2};

  dim3 gE((MROWS_c + 127) / 128, (E_c + 127) / 128);
  dim3 gF((MROWS_c + 127) / 128, (F_c + 127) / 128);
  unsigned gridE  = (unsigned)(((size_t)MROWS_c * E_c + 255) / 256);
  unsigned gEE    = (unsigned)(((size_t)E_c * E_c + 255) / 256);
  unsigned gEF    = (unsigned)(((size_t)E_c * F_c + 255) / 256);

  for (int l = 0; l < L_c; ++l) {
    size_t we = (size_t)l * E_c * E_c;
    size_t wf = (size_t)l * E_c * F_c;
    size_t be = (size_t)l * E_c;
    size_t bf = (size_t)l * F_c;

    // pre-convert this layer's weights to bf16 [N,K]
    wconv_t_kernel<<<gEE, 256, 0, stream>>>(wq + we, wtq, E_c, E_c);
    wconv_t_kernel<<<gEE, 256, 0, stream>>>(wk + we, wtk, E_c, E_c);
    wconv_t_kernel<<<gEE, 256, 0, stream>>>(wv + we, wtv, E_c, E_c);
    wconv_t_kernel<<<gEE, 256, 0, stream>>>(wo + we, wto, E_c, E_c);
    wconv_t_kernel<<<gEF, 256, 0, stream>>>(w1 + wf, wt1, E_c, F_c);  // -> [F][E]
    wconv_t_kernel<<<gEF, 256, 0, stream>>>(w2 + wf, wt2, F_c, E_c);  // -> [E][F]

    layernorm_kernel<<<MROWS_c, 256, 0, stream>>>(h, ln1_g + be, ln1_b + be, y);
    gemm_kernel<1><<<gE, 256, 0, stream>>>(y, wtq, bq + be, nullptr, qb, MROWS_c, E_c, E_c);
    gemm_kernel<1><<<gE, 256, 0, stream>>>(y, wtk, bk + be, nullptr, kb, MROWS_c, E_c, E_c);
    gemm_kernel<1><<<gE, 256, 0, stream>>>(y, wtv, bv + be, nullptr, vb, MROWS_c, E_c, E_c);

    for (int br = 0; br < 3; ++br) {
      int blocks = B_c * nsg[br] * NH_c * 8;
      attn_branch_kernel<<<blocks, 128, 0, stream>>>(qb, kb, vb, obrs[br], lbrs[br],
                                                     wsz[br], rsz[br], nsg[br]);
    }
    combine_kernel<<<gridE, 256, 0, stream>>>(o0, o1, o2, l0, l1, l2, ao);

    gemm_kernel<5><<<gE, 256, 0, stream>>>(ao, wto, bo + be, h, h, MROWS_c, E_c, E_c);
    layernorm_kernel<<<MROWS_c, 256, 0, stream>>>(h, ln2_g + be, ln2_b + be, y);
    gemm_kernel<3><<<gF, 256, 0, stream>>>(y, wt1, b1 + bf, nullptr, mid, MROWS_c, F_c, E_c);
    gemm_kernel<5><<<gE, 256, 0, stream>>>(mid, wt2, b2 + be, h, h, MROWS_c, E_c, F_c);
  }

  layernorm_kernel<<<MROWS_c, 256, 0, stream>>>(h, nf_g, nf_b, y);
  extract_kernel<<<(B_c * E_c + 255) / 256, 256, 0, stream>>>(y, out);
}
